// DeepSeekMoE_35845797052871
// MI455X (gfx1250) — matrix operand order repaired and checked by reading, not compiled, not run on hardware
//
#include <hip/hip_runtime.h>
#include <hip/hip_bf16.h>

typedef __attribute__((ext_vector_type(16))) __bf16 v16bf;
typedef __attribute__((ext_vector_type(8)))  __bf16 v8bf;
typedef __attribute__((ext_vector_type(8)))  float  v8f;

#define N_TOK 32768
#define CDIM  1024
#define EXP   8
#define IDIM  128
#define ISDIM 256
#define NUP   1536   // ISDIM + 10*IDIM
#define KTOP  6

// ---------------- fragment loads (wave32 WMMA bf16 layouts) ----------------

// A fragment: 16x32 bf16, A row-major [M,K]. lane: m=L&15, hi=L>>4
// holds K = hi*16 .. hi*16+15 contiguous
__device__ __forceinline__ v16bf load_a_frag(const __bf16* __restrict__ A, int lda,
                                             int row0, int k0, int lane) {
    const int m = lane & 15, hi = lane >> 4;
    const __bf16* p = A + (size_t)(row0 + m) * lda + k0 + hi * 16;
    return *(const v16bf*)p;
}

// B fragment: 32x16 bf16 from transposed weights Bt[n_out][K].
// lane: n=L&15 fixed, holds K = hi*16 .. hi*16+15 contiguous
__device__ __forceinline__ v16bf load_b_frag(const __bf16* __restrict__ Bt, int ldb,
                                             int col0, int k0, int lane) {
    const int n = lane & 15, hi = lane >> 4;
    return *(const v16bf*)(Bt + (size_t)(col0 + n) * ldb + k0 + hi * 16);
}

// ---------------- prep kernels ----------------

__global__ __launch_bounds__(256) void k_cvt_x(const float* __restrict__ x,
                                               __bf16* __restrict__ Xb, int n) {
    int stride = gridDim.x * blockDim.x * 4;
    for (int i = (blockIdx.x * blockDim.x + threadIdx.x) * 4; i < n; i += stride) {
        float4 v = *(const float4*)(x + i);
        Xb[i + 0] = (__bf16)v.x;
        Xb[i + 1] = (__bf16)v.y;
        Xb[i + 2] = (__bf16)v.z;
        Xb[i + 3] = (__bf16)v.w;
    }
}

// Build transposed bf16 weights:
//  Gt[j][c] = (j<256 ? ws1[c][j] : w1[e][c][jj]),  Ut likewise from ws2/w2
//  Dt[c][j] = (j<256 ? ws3[j][c] : w3[e][jj][c])
__global__ __launch_bounds__(256) void k_prep_w(const float* __restrict__ w1,
                                                const float* __restrict__ w2,
                                                const float* __restrict__ w3,
                                                const float* __restrict__ ws1,
                                                const float* __restrict__ ws2,
                                                const float* __restrict__ ws3,
                                                __bf16* __restrict__ Gt,
                                                __bf16* __restrict__ Ut,
                                                __bf16* __restrict__ Dt) {
    const int TOT = NUP * CDIM;
    int stride = gridDim.x * blockDim.x;
    for (int i = blockIdx.x * blockDim.x + threadIdx.x; i < 3 * TOT; i += stride) {
        int which = i / TOT, r = i - which * TOT;
        if (which < 2) {
            int j = r >> 10, c = r & 1023;           // [j][c], ld = 1024
            const float* wsX = which ? ws2 : ws1;
            const float* wX  = which ? w2  : w1;
            float v;
            if (j < ISDIM) v = wsX[(size_t)c * ISDIM + j];
            else {
                int e = (j - ISDIM) >> 7, jj = (j - ISDIM) & 127;
                v = wX[((size_t)e * CDIM + c) * IDIM + jj];
            }
            (which ? Ut : Gt)[r] = (__bf16)v;
        } else {
            int c = r / NUP, j = r - c * NUP;        // [c][j], ld = 1536
            float v;
            if (j < ISDIM) v = ws3[(size_t)j * CDIM + c];
            else {
                int e = (j - ISDIM) >> 7, jj = (j - ISDIM) & 127;
                v = w3[((size_t)e * IDIM + jj) * CDIM + c];
            }
            Dt[r] = (__bf16)v;
        }
    }
}

// Router: one wave per token. logits = x[n]·wr -> softmax -> keep top-6 -> renorm.
__global__ __launch_bounds__(256) void k_router(const float* __restrict__ x,
                                                const float* __restrict__ wr,
                                                float* __restrict__ scale) {
    const int lane = threadIdx.x & 31;
    const int n = blockIdx.x * 8 + (threadIdx.x >> 5);
    float acc[EXP];
#pragma unroll
    for (int e = 0; e < EXP; ++e) acc[e] = 0.0f;
    const float* xrow = x + (size_t)n * CDIM;
    for (int c = lane; c < CDIM; c += 32) {
        float xv = xrow[c];
        const float* wrow = wr + (size_t)c * EXP;
#pragma unroll
        for (int e = 0; e < EXP; ++e) acc[e] += xv * wrow[e];
    }
#pragma unroll
    for (int e = 0; e < EXP; ++e)
#pragma unroll
        for (int off = 16; off > 0; off >>= 1)
            acc[e] += __shfl_xor(acc[e], off, 32);
    if (lane == 0) {
        float mx = acc[0];
#pragma unroll
        for (int e = 1; e < EXP; ++e) mx = fmaxf(mx, acc[e]);
        float p[EXP], sum = 0.0f;
#pragma unroll
        for (int e = 0; e < EXP; ++e) { p[e] = __expf(acc[e] - mx); sum += p[e]; }
        float inv = 1.0f / sum;
#pragma unroll
        for (int e = 0; e < EXP; ++e) p[e] *= inv;
        // drop the (E - K) smallest
        for (int d = 0; d < EXP - KTOP; ++d) {
            int mi = -1; float mv = 1e30f;
#pragma unroll
            for (int e = 0; e < EXP; ++e)
                if (p[e] >= 0.0f && p[e] < mv) { mv = p[e]; mi = e; }
            p[mi] = -1.0f;
        }
        float s2 = 0.0f;
#pragma unroll
        for (int e = 0; e < EXP; ++e) if (p[e] > 0.0f) s2 += p[e];
        float r = 1.0f / s2;
#pragma unroll
        for (int e = 0; e < EXP; ++e)
            scale[(size_t)n * EXP + e] = (p[e] > 0.0f) ? p[e] * r : 0.0f;
    }
}

// ---------------- up: H = silu(X*Gt^T) .* (X*Ut^T) .* s, bf16 out ----------------
// block: 256 thr = 8 waves as 2(M)x4(N); wave tile 32(M)x16(N); block 64x64.
__global__ __launch_bounds__(256) void k_up(const __bf16* __restrict__ Xb,
                                            const __bf16* __restrict__ Gt,
                                            const __bf16* __restrict__ Ut,
                                            const float* __restrict__ scale,
                                            __bf16* __restrict__ Hb) {
    const int lane = threadIdx.x & 31;
    const int w = threadIdx.x >> 5;
    const int wn = w & 3, wm = w >> 2;
    const int row0 = blockIdx.y * 64 + wm * 32;
    const int col0 = blockIdx.x * 64 + wn * 16;

    v8f cg[2] = {}, cu[2] = {};
#pragma unroll 4
    for (int k0 = 0; k0 < CDIM; k0 += 32) {
        if (k0 + 32 < CDIM)
            __builtin_prefetch(Xb + (size_t)(row0 + (lane & 15)) * CDIM + k0 + 32, 0, 1);
        v16bf a0 = load_a_frag(Xb, CDIM, row0,      k0, lane);
        v16bf a1 = load_a_frag(Xb, CDIM, row0 + 16, k0, lane);
        v16bf bg = load_b_frag(Gt, CDIM, col0, k0, lane);
        v16bf bu = load_b_frag(Ut, CDIM, col0, k0, lane);
        cg[0] = __builtin_amdgcn_wmma_f32_16x16x32_bf16(false, a0, false, bg, (short)0, cg[0], false, false);
        cg[1] = __builtin_amdgcn_wmma_f32_16x16x32_bf16(false, a1, false, bg, (short)0, cg[1], false, false);
        cu[0] = __builtin_amdgcn_wmma_f32_16x16x32_bf16(false, a0, false, bu, (short)0, cu[0], false, false);
        cu[1] = __builtin_amdgcn_wmma_f32_16x16x32_bf16(false, a1, false, bu, (short)0, cu[1], false, false);
    }

    const int e = (col0 < ISDIM) ? -1 : ((col0 - ISDIM) >> 7);
    const int nc = lane & 15, hi = lane >> 4;
#pragma unroll
    for (int t = 0; t < 2; ++t) {
#pragma unroll
        for (int r = 0; r < 8; ++r) {
            int m = row0 + t * 16 + hi * 8 + r;
            float g = cg[t][r], u = cu[t][r];
            float s = (e < 0) ? 1.0f : scale[(size_t)m * EXP + e];
            float h = (g / (1.0f + __expf(-g))) * u * s;
            Hb[(size_t)m * NUP + col0 + nc] = (__bf16)h;
        }
    }
}

// ---------------- down: out = H * Dt^T, fp32 out ----------------
__global__ __launch_bounds__(256) void k_down(const __bf16* __restrict__ Hb,
                                              const __bf16* __restrict__ Dt,
                                              float* __restrict__ out) {
    const int lane = threadIdx.x & 31;
    const int w = threadIdx.x >> 5;
    const int wn = w & 3, wm = w >> 2;
    const int row0 = blockIdx.y * 64 + wm * 32;
    const int col0 = blockIdx.x * 64 + wn * 16;

    v8f cd[2] = {};
#pragma unroll 4
    for (int k0 = 0; k0 < NUP; k0 += 32) {
        if (k0 + 32 < NUP)
            __builtin_prefetch(Hb + (size_t)(row0 + (lane & 15)) * NUP + k0 + 32, 0, 1);
        v16bf a0 = load_a_frag(Hb, NUP, row0,      k0, lane);
        v16bf a1 = load_a_frag(Hb, NUP, row0 + 16, k0, lane);
        v16bf bd = load_b_frag(Dt, NUP, col0, k0, lane);
        cd[0] = __builtin_amdgcn_wmma_f32_16x16x32_bf16(false, a0, false, bd, (short)0, cd[0], false, false);
        cd[1] = __builtin_amdgcn_wmma_f32_16x16x32_bf16(false, a1, false, bd, (short)0, cd[1], false, false);
    }

    const int nc = lane & 15, hi = lane >> 4;
#pragma unroll
    for (int t = 0; t < 2; ++t) {
#pragma unroll
        for (int r = 0; r < 8; ++r) {
            int m = row0 + t * 16 + hi * 8 + r;
            out[(size_t)m * CDIM + col0 + nc] = cd[t][r];
        }
    }
}

// ---------------- launch ----------------

extern "C" void kernel_launch(void* const* d_in, const int* in_sizes, int n_in,
                              void* d_out, int out_size, void* d_ws, size_t ws_size,
                              hipStream_t stream) {
    const float* x   = (const float*)d_in[0];
    const float* w1  = (const float*)d_in[1];
    const float* w2  = (const float*)d_in[2];
    const float* w3  = (const float*)d_in[3];
    const float* ws1 = (const float*)d_in[4];
    const float* ws2 = (const float*)d_in[5];
    const float* ws3 = (const float*)d_in[6];
    const float* wr  = (const float*)d_in[7];
    float* out = (float*)d_out;

    char* wsp = (char*)d_ws;
    // workspace layout (bytes, all 256-aligned)
    __bf16* Xb    = (__bf16*)(wsp);                       // 32768*1024*2  = 64 MiB
    __bf16* Gt    = (__bf16*)(wsp + 67108864);            // 1536*1024*2   =  3 MiB
    __bf16* Ut    = (__bf16*)(wsp + 70254592);            //               =  3 MiB
    __bf16* Dt    = (__bf16*)(wsp + 73400320);            // 1024*1536*2   =  3 MiB
    float*  scale = (float*) (wsp + 76546048);            // 32768*8*4     =  1 MiB
    __bf16* Hb    = (__bf16*)(wsp + 77594624);            // 32768*1536*2  = 96 MiB

    const int nX = N_TOK * CDIM;
    k_cvt_x <<<nX / (256 * 4), 256, 0, stream>>>(x, Xb, nX);
    k_prep_w<<<4608, 256, 0, stream>>>(w1, w2, w3, ws1, ws2, ws3, Gt, Ut, Dt);
    k_router<<<N_TOK / 8, 256, 0, stream>>>(x, wr, scale);
    k_up   <<<dim3(NUP / 64,  N_TOK / 64), 256, 0, stream>>>(Xb, Gt, Ut, scale, Hb);
    k_down <<<dim3(CDIM / 64, N_TOK / 64), 256, 0, stream>>>(Hb, Dt, out);
}
